// CNN_OTAM_18648747999712
// MI455X (gfx1250) — matrix operand order repaired and checked by reading, compiled, not run on hardware
//
#include <hip/hip_runtime.h>
#include <hip/hip_fp16.h>

typedef _Float16 v16h __attribute__((ext_vector_type(16)));
typedef _Float16 v8h  __attribute__((ext_vector_type(8)));
typedef float    v8f  __attribute__((ext_vector_type(8)));

#define OTAM_LBDA   0.1f
#define OTAM_INVL   10.0f
#define DIM 2048
#define SEQ 16

// ---------------------------------------------------------------------------
// Kernel 1: L2-normalize each 2048-dim frame row (fp32 -> fp16)
// One 256-thread block per row; 8 elements / thread.
// ---------------------------------------------------------------------------
__global__ __launch_bounds__(256)
void otam_normalize_rows(const float* __restrict__ tf, const float* __restrict__ sf,
                         _Float16* __restrict__ tfn, _Float16* __restrict__ sfn,
                         int n_t_rows) {
    __shared__ float red[256];
    int row = blockIdx.x;
    const float* src;
    _Float16* dst;
    if (row < n_t_rows) {
        src = tf + (size_t)row * DIM;
        dst = tfn + (size_t)row * DIM;
    } else {
        int r = row - n_t_rows;
        src = sf + (size_t)r * DIM;
        dst = sfn + (size_t)r * DIM;
    }
    int t = threadIdx.x;
    const float4* s4 = (const float4*)src;
    float4 a = s4[t * 2 + 0];
    float4 b = s4[t * 2 + 1];
    float ss = a.x*a.x + a.y*a.y + a.z*a.z + a.w*a.w
             + b.x*b.x + b.y*b.y + b.z*b.z + b.w*b.w;
    red[t] = ss;
    __syncthreads();
    #pragma unroll
    for (int off = 128; off > 0; off >>= 1) {
        if (t < off) red[t] += red[t + off];
        __syncthreads();
    }
    float scale = 1.0f / fmaxf(sqrtf(red[0]), 1e-8f);
    v8h h;
    h[0] = (_Float16)(a.x * scale); h[1] = (_Float16)(a.y * scale);
    h[2] = (_Float16)(a.z * scale); h[3] = (_Float16)(a.w * scale);
    h[4] = (_Float16)(b.x * scale); h[5] = (_Float16)(b.y * scale);
    h[6] = (_Float16)(b.z * scale); h[7] = (_Float16)(b.w * scale);
    *(v8h*)(dst + (size_t)t * 8) = h;
}

// ---------------------------------------------------------------------------
// Kernel 2: zero the class accumulator
// ---------------------------------------------------------------------------
__global__ void otam_zero_f32(float* __restrict__ p, int n) {
    int i = blockIdx.x * blockDim.x + threadIdx.x;
    if (i < n) p[i] = 0.0f;
}

// ---------------------------------------------------------------------------
// Kernel 3: fused WMMA cosine-distance tile + OTAM soft-DTW (both directions)
// One wave (32 lanes) per (query, support) pair. 8 waves / block.
// ---------------------------------------------------------------------------
__global__ __launch_bounds__(256)
void otam_wmma_dtw(const _Float16* __restrict__ tfn,
                   const _Float16* __restrict__ sfn,
                   const int* __restrict__ labels,
                   float* __restrict__ acc,
                   int nQ, int nS, int C) {
    __shared__ float tiles[8 * 256];
    const int lane = threadIdx.x & 31;
    const int wave = threadIdx.x >> 5;
    long long pair = (long long)blockIdx.x * 8 + wave;
    if (pair >= (long long)nQ * nS) return;
    const int q = (int)(pair / nS);
    const int s = (int)(pair % nS);

    const _Float16* aBase = tfn + (size_t)q * SEQ * DIM;
    const _Float16* bBase = sfn + (size_t)s * SEQ * DIM;
    const int hi16 = (lane >= 16);
    const int mrow = lane & 15;

    // A operand (16x32 f16): lanes 0-15 hold row m, K chunks {0..7,16..23};
    //                        lanes 16-31 hold row m, K chunks {8..15,24..31}.
    const _Float16* aRow = aBase + (size_t)mrow * DIM + (hi16 ? 8 : 0);
    // B operand (32x16 f16): lane n holds column n; lanes 0-15 K chunks {0..7,16..23};
    //                        lanes 16-31 K chunks {8..15,24..31}.
    const _Float16* bRow = bBase + (size_t)mrow * DIM + (hi16 ? 8 : 0);

    v8f c = {0.f, 0.f, 0.f, 0.f, 0.f, 0.f, 0.f, 0.f};
    #pragma unroll 4
    for (int k0 = 0; k0 < DIM; k0 += 32) {
        v8h alo = *(const v8h*)(aRow + k0);
        v8h ahi = *(const v8h*)(aRow + k0 + 16);
        v8h blo = *(const v8h*)(bRow + k0);
        v8h bhi = *(const v8h*)(bRow + k0 + 16);
        v16h av = __builtin_shufflevector(alo, ahi, 0,1,2,3,4,5,6,7,8,9,10,11,12,13,14,15);
        v16h bv = __builtin_shufflevector(blo, bhi, 0,1,2,3,4,5,6,7,8,9,10,11,12,13,14,15);
        c = __builtin_amdgcn_wmma_f32_16x16x32_f16(
                /*neg_a=*/false, av, /*neg_b=*/false, bv,
                /*c_mod=*/(short)0, c, /*reuse_a=*/false, /*reuse_b=*/false);
    }

    // Scatter distance tile (1 - cossim) to LDS.
    // C/D layout: VGPR r holds M=r (lanes 0-15) / M=r+8 (lanes 16-31), N=lane%16.
    float* tile = &tiles[wave * 256];
    #pragma unroll
    for (int r = 0; r < 8; ++r) {
        int m = r + (hi16 ? 8 : 0);
        tile[m * 16 + mrow] = 1.0f - c[r];
    }
    // Fence LDS writes before cross-lane reads (same-wave DS ordering + wait).
    __asm__ volatile("s_wait_dscnt 0" ::: "memory");

    // OTAM soft-DTW: lane 0 = forward grid [Lq, Ms], lane 1 = transposed grid.
    if (lane < 2) {
        const int dir = lane;
        float prev[18], cur[18];
        // Row 0: cumulative sum of padded distances (pad cols 0 and 17 are 0).
        prev[0] = 0.0f;
        #pragma unroll
        for (int m = 1; m <= 16; ++m) {
            float d0 = dir ? tile[(m - 1) * 16 + 0] : tile[0 * 16 + (m - 1)];
            prev[m] = prev[m - 1] + d0;
        }
        prev[17] = prev[16];

        for (int l = 1; l < 16; ++l) {
            cur[0] = 0.0f;
            #pragma unroll
            for (int m = 1; m <= 17; ++m) {
                float dm = (m <= 16)
                    ? (dir ? tile[(m - 1) * 16 + l] : tile[l * 16 + (m - 1)])
                    : 0.0f;
                float c0 = prev[m - 1];   // cum[l-1, m-1]
                float c1 = cur[m - 1];    // cum[l,   m-1]
                float mn = fminf(c0, c1);
                float ssum = __expf((mn - c0) * OTAM_INVL)
                           + __expf((mn - c1) * OTAM_INVL);
                if (m == 1 || m == 17) {  // boundary: include cum[l-1, m]
                    float c2 = prev[m];
                    float mn2 = fminf(mn, c2);
                    if (mn2 < mn) { ssum *= __expf((mn2 - mn) * OTAM_INVL); mn = mn2; }
                    ssum += __expf((mn - c2) * OTAM_INVL);
                }
                cur[m] = dm + mn - OTAM_LBDA * __logf(ssum);
            }
            #pragma unroll
            for (int m = 0; m < 18; ++m) prev[m] = cur[m];
        }
        // Both lanes (fwd + transposed) accumulate into the class slot.
        atomicAdd(&acc[(size_t)q * C + labels[s]], prev[17]);
    }
}

// ---------------------------------------------------------------------------
// Kernel 4: out[q, c] = -acc[q, c] / count(c)
// ---------------------------------------------------------------------------
__global__ void otam_finalize(const float* __restrict__ acc,
                              const int* __restrict__ labels,
                              float* __restrict__ out,
                              int nQ, int C, int nS) {
    int i = blockIdx.x * blockDim.x + threadIdx.x;
    if (i >= nQ * C) return;
    int cls = i % C;
    int cnt = 0;
    for (int s = 0; s < nS; ++s) cnt += (labels[s] == cls) ? 1 : 0;
    if (cnt < 1) cnt = 1;
    out[i] = -acc[i] / (float)cnt;
}

// ---------------------------------------------------------------------------
extern "C" void kernel_launch(void* const* d_in, const int* in_sizes, int n_in,
                              void* d_out, int out_size, void* d_ws, size_t ws_size,
                              hipStream_t stream) {
    const float* sf     = (const float*)d_in[0];   // support [S, L, D]
    const float* tf     = (const float*)d_in[1];   // target  [Q, L, D]
    const int*   labels = (const int*)d_in[2];     // [S]

    const int nS = in_sizes[2];                    // 25
    const int nQ = in_sizes[1] / (SEQ * DIM);      // 2000
    const int C  = out_size / nQ;                  // 5

    // Workspace layout: f16 target copy | f16 support copy | f32 class acc
    size_t tHalfs = (size_t)nQ * SEQ * DIM;
    size_t sHalfs = (size_t)nS * SEQ * DIM;
    _Float16* tfn = (_Float16*)d_ws;
    _Float16* sfn = tfn + tHalfs;
    float*    acc = (float*)(sfn + sHalfs);

    const int tRows = nQ * SEQ, sRows = nS * SEQ;
    otam_normalize_rows<<<tRows + sRows, 256, 0, stream>>>(tf, sf, tfn, sfn, tRows);

    const int accN = nQ * C;
    otam_zero_f32<<<(accN + 255) / 256, 256, 0, stream>>>(acc, accN);

    const long long pairs = (long long)nQ * nS;
    const int blocks = (int)((pairs + 7) / 8);
    otam_wmma_dtw<<<blocks, 256, 0, stream>>>(tfn, sfn, labels, acc, nQ, nS, C);

    otam_finalize<<<(accN + 255) / 256, 256, 0, stream>>>(acc, labels, (float*)d_out, nQ, C, nS);
}
